// Attention_9586367005194
// MI455X (gfx1250) — matrix operand order repaired and checked by reading, not compiled, not run on hardware
//
#include <hip/hip_runtime.h>
#include <hip/hip_bf16.h>

#define DIMN 4096
#define NH   32
#define NKV  8
#define HD   128
#define SEQ  2048

typedef __attribute__((ext_vector_type(16))) __bf16        bf16x16;
typedef __attribute__((ext_vector_type(8)))  float         f32x8;
typedef __attribute__((ext_vector_type(4))) unsigned int   u32x4;
typedef __attribute__((ext_vector_type(8)))  int           i32x8;
typedef __attribute__((ext_vector_type(4)))  int           i32x4;

union FragAB { bf16x16 v; u32x4 q[2]; };

__device__ __forceinline__ unsigned short f32_to_bf16(float f) {
  unsigned int u = __builtin_bit_cast(unsigned int, f);
  u += 0x7FFFu + ((u >> 16) & 1u);           // round-to-nearest-even
  return (unsigned short)(u >> 16);
}

// A fragment 16x32 bf16 (row-major source, ld elements per row), tile K-origin k0.
// ISA 7.12.2: lanes 0-15 -> M=lane, K 0..7 then 16..23; lanes 16-31 -> K 8..15 then 24..31.
__device__ __forceinline__ FragAB load_frag_a(const unsigned short* base, int ld, int k0, int lane) {
  const int lh = lane & 15, hf = lane >> 4;
  const unsigned short* p = base + (size_t)lh * ld + k0 + 8 * hf;
  FragAB f;
  f.q[0] = *(const u32x4*)(p);
  f.q[1] = *(const u32x4*)(p + 16);
  return f;
}

// B fragment 32x16 bf16 from an N-major / K-contiguous source (B^T row-major).
// Lane holds column n = lane&15; lanes 0-15 hold K 0..7 then 16..23, lanes 16-31 hold K 8..15 then 24..31.
// Works for both global and LDS (generic) pointers.
__device__ __forceinline__ FragAB load_frag_b(const unsigned short* base, int ld, int k0, int lane) {
  const int lh = lane & 15, hf = lane >> 4;
  const unsigned short* p = base + (size_t)(lh)*ld + k0 + 8 * hf;
  FragAB f;
  f.q[0] = *(const u32x4*)(p);
  f.q[1] = *(const u32x4*)(p + 16);
  return f;
}

__device__ __forceinline__ f32x8 wmma_bf16(const FragAB& a, const FragAB& b, f32x8 c) {
  return __builtin_amdgcn_wmma_f32_16x16x32_bf16(false, a.v, false, b.v, (short)0, c, false, false);
}

// ---- Tensor Data Mover: 2-D tile (rows of tile_d0 elems, tile_d1 rows, 2B elems) -> LDS ----
// D# per CDNA5 ISA.  Groups 2/3 zero (<=2D tensor).  Tracked by TENSORcnt.
__device__ __forceinline__ void tdm_load_2d_to_lds(void* lds_dst, const void* gsrc,
                                                   unsigned tile_d0, unsigned tile_d1,
                                                   unsigned long long row_stride_elems) {
  const unsigned lds_off = (unsigned)(unsigned long long)lds_dst;    // low 32b of generic = LDS byte offset
  const unsigned long long ga = (unsigned long long)gsrc;
  u32x4 g0;
  g0[0] = 1u;                                                        // count=1, user mode, no gather
  g0[1] = lds_off;                                                   // lds_addr
  g0[2] = (unsigned)(ga & 0xffffffffu);                              // global_addr[31:0]
  g0[3] = (unsigned)((ga >> 32) & 0x01ffffffu) | (2u << 30);         // global_addr[56:32] | type=2
  i32x8 g1;
  g1[0] = (int)(1u << 16);                                           // data_size=1 (2 bytes), wg_mask=0
  g1[1] = (int)((tile_d0 & 0xffffu) << 16);                          // tensor_dim0[15:0]
  g1[2] = (int)(((tile_d0 >> 16) & 0xffffu) | ((tile_d1 & 0xffffu) << 16)); // td0 hi | td1 lo
  g1[3] = (int)(((tile_d1 >> 16) & 0xffffu) | ((tile_d0 & 0xffffu) << 16)); // td1 hi | tile_dim0
  g1[4] = (int)(tile_d1 & 0xffffu);                                  // tile_dim1 (tile_dim2=0)
  g1[5] = (int)(row_stride_elems & 0xffffffffu);                     // tensor_dim0_stride lo
  g1[6] = (int)((row_stride_elems >> 32) & 0xffffu);                 // stride hi (dim1_stride=0)
  g1[7] = 0;
  i32x4 z4 = {0, 0, 0, 0};
#if __clang_major__ >= 23
  i32x8 z8 = {0, 0, 0, 0, 0, 0, 0, 0};
  __builtin_amdgcn_tensor_load_to_lds(g0, g1, z4, z4, z8, 0);
#else
  __builtin_amdgcn_tensor_load_to_lds(g0, g1, z4, z4, 0);
#endif
}

// ---------------- elementwise converts ----------------

__global__ void cvt_f32_bf16(const float* __restrict__ in, unsigned short* __restrict__ out, int n) {
  int i = blockIdx.x * 256 + threadIdx.x;
  if (i < n) out[i] = f32_to_bf16(in[i]);
}

// in: [K][N] f32  ->  out: [N][K] bf16
__global__ void transpose_cvt(const float* __restrict__ in, unsigned short* __restrict__ out, int K, int N) {
  int idx = blockIdx.x * 256 + threadIdx.x;
  if (idx < K * N) {
    int k = idx / N, n = idx % N;
    out[(size_t)n * K + k] = f32_to_bf16(in[idx]);
  }
}

// RoPE (interleaved pairs) + bf16 convert.  in/out: [SEQ][nheads][HD]
__global__ void rope_cvt(const float* __restrict__ in, const float* __restrict__ fc,
                         const float* __restrict__ fs, unsigned short* __restrict__ out, int nheads) {
  int idx = blockIdx.x * 256 + threadIdx.x;
  int total = SEQ * nheads * (HD / 2);
  if (idx >= total) return;
  int i = idx % (HD / 2);
  int h = (idx / (HD / 2)) % nheads;
  int s = idx / ((HD / 2) * nheads);
  float c  = fc[s * (HD / 2) + i];
  float sn = fs[s * (HD / 2) + i];
  size_t base = ((size_t)s * nheads + h) * HD + 2 * i;
  float tr = in[base], ti = in[base + 1];
  out[base]     = f32_to_bf16(tr * c - ti * sn);
  out[base + 1] = f32_to_bf16(tr * sn + ti * c);
}

// xv: [SEQ][NKV][HD] f32  ->  vT: [NKV][HD][SEQ] bf16
__global__ void v_transpose_cvt(const float* __restrict__ xv, unsigned short* __restrict__ vT) {
  int idx = blockIdx.x * 256 + threadIdx.x;
  if (idx >= NKV * HD * SEQ) return;
  int s = idx % SEQ;
  int d = (idx / SEQ) % HD;
  int h = idx / (SEQ * HD);
  vT[idx] = f32_to_bf16(xv[((size_t)s * NKV + h) * HD + d]);
}

// ---------------- software-pipelined bf16 WMMA GEMM ----------------
// C(f32, MxN) = A(bf16, MxK) * B^T(bf16, NxK).  Block = 4 waves, 128x64 C tile;
// wave computes 32x64 with next K-chunk fragments loaded ahead of the current WMMAs.

__global__ __launch_bounds__(128) void gemm_bf16(const unsigned short* __restrict__ A,
                                                 const unsigned short* __restrict__ BT,
                                                 float* __restrict__ C, int M, int N, int K) {
  const int lane = threadIdx.x & 31;
  const int wave = threadIdx.x >> 5;
  const int m0 = blockIdx.y * 128 + wave * 32;
  const int n0 = blockIdx.x * 64;

  f32x8 cf[2][4];
  #pragma unroll
  for (int r = 0; r < 2; ++r)
    #pragma unroll
    for (int t = 0; t < 4; ++t)
      #pragma unroll
      for (int i = 0; i < 8; ++i) cf[r][t][i] = 0.0f;

  const unsigned short* ab0 = A + (size_t)m0 * K;
  const unsigned short* ab1 = A + (size_t)(m0 + 16) * K;

  FragAB a0 = load_frag_a(ab0, K, 0, lane);
  FragAB a1 = load_frag_a(ab1, K, 0, lane);
  FragAB b[4];
  #pragma unroll
  for (int t = 0; t < 4; ++t) b[t] = load_frag_b(BT + (size_t)(n0 + t * 16) * K, K, 0, lane);

  for (int k0 = 0; k0 < K - 32; k0 += 32) {
    const int kn = k0 + 32;
    __builtin_prefetch(ab0 + (size_t)(lane & 15) * K + kn + 32, 0, 1);
    __builtin_prefetch(ab1 + (size_t)(lane & 15) * K + kn + 32, 0, 1);
    FragAB a0n = load_frag_a(ab0, K, kn, lane);
    FragAB a1n = load_frag_a(ab1, K, kn, lane);
    FragAB bn[4];
    #pragma unroll
    for (int t = 0; t < 4; ++t) bn[t] = load_frag_b(BT + (size_t)(n0 + t * 16) * K, K, kn, lane);

    #pragma unroll
    for (int t = 0; t < 4; ++t) {
      cf[0][t] = wmma_bf16(a0, b[t], cf[0][t]);
      cf[1][t] = wmma_bf16(a1, b[t], cf[1][t]);
    }
    a0 = a0n; a1 = a1n;
    #pragma unroll
    for (int t = 0; t < 4; ++t) b[t] = bn[t];
  }
  #pragma unroll
  for (int t = 0; t < 4; ++t) {
    cf[0][t] = wmma_bf16(a0, b[t], cf[0][t]);
    cf[1][t] = wmma_bf16(a1, b[t], cf[1][t]);
  }

  const int lh = lane & 15, hf = lane >> 4;
  #pragma unroll
  for (int r = 0; r < 2; ++r)
    #pragma unroll
    for (int t = 0; t < 4; ++t)
      #pragma unroll
      for (int i = 0; i < 8; ++i)
        C[(size_t)(m0 + r * 16 + i + 8 * hf) * N + n0 + t * 16 + lh] = cf[r][t][i];
}

// ---------------- flash attention (causal, GQA) with TDM-staged K/V ----------------
// q: [SEQ][NH][HD] bf16, k: [SEQ][NKV][HD] bf16, vT: [NKV][HD][SEQ] bf16, o: [SEQ][NH][HD] bf16
// Block = 4 waves = 64 query rows, one wave per 16 rows; grid = (SEQ/64, NH).
// Wave 0 drives the Tensor Data Mover: K (32x128) and V^T (128x32) tiles for block jb+1
// stream into the opposite LDS buffer while all waves consume block jb from LDS.

__global__ __launch_bounds__(128) void flash_attn(const unsigned short* __restrict__ q,
                                                  const unsigned short* __restrict__ k,
                                                  const unsigned short* __restrict__ vT,
                                                  unsigned short* __restrict__ o) {
  __shared__ unsigned short kt[2][32 * 128];     // K tiles: 32 keys x 128 dims
  __shared__ unsigned short vt[2][128 * 32];     // V^T tiles: 128 dims x 32 keys
  __shared__ unsigned short lds_p[4][16 * 32];   // per-wave P staging (C->A transpose)

  const int lane = threadIdx.x & 31;
  const int wave = threadIdx.x >> 5;
  const int head = blockIdx.y;
  const int kvh  = head >> 2;                    // N_REP = 4
  const int qb   = blockIdx.x * 64;
  const int q0   = qb + wave * 16;
  const int lh = lane & 15, hf = lane >> 4;
  const float scale = 0.08838834764831845f;      // 1/sqrt(128)

  FragAB qa[4];
  const unsigned short* qbase = q + (size_t)q0 * (NH * HD) + head * HD;
  #pragma unroll
  for (int c = 0; c < 4; ++c) qa[c] = load_frag_a(qbase, NH * HD, c * 32, lane);

  f32x8 of[8];
  #pragma unroll
  for (int t = 0; t < 8; ++t)
    #pragma unroll
    for (int i = 0; i < 8; ++i) of[t][i] = 0.0f;

  float Mx[8], Lx[8];
  #pragma unroll
  for (int i = 0; i < 8; ++i) { Mx[i] = -__builtin_inff(); Lx[i] = 0.0f; }

  const unsigned short* kbase = k + (size_t)kvh * HD;                 // row stride NKV*HD
  const unsigned short* vbase = vT + (size_t)kvh * HD * SEQ;          // [HD][SEQ]
  unsigned short* pl = lds_p[wave];

  const int nblocks = (qb + 63) / 32 + 1;        // block-uniform causal extent

  // prologue: stage block 0 via TDM
  if (wave == 0) {
    tdm_load_2d_to_lds(&kt[0][0], kbase, HD, 32, (unsigned long long)(NKV * HD));
    tdm_load_2d_to_lds(&vt[0][0], vbase, 32, HD, (unsigned long long)SEQ);
  }

  for (int jb = 0; jb < nblocks; ++jb) {
    const int j = jb * 32;
    if (wave == 0) __builtin_amdgcn_s_wait_tensorcnt(0);   // buffer jb landed in LDS
    __syncthreads();                                       // publish to all waves
    if (wave == 0 && jb + 1 < nblocks) {                   // stream next block during compute
      const int jn = j + 32;
      tdm_load_2d_to_lds(&kt[(jb + 1) & 1][0], kbase + (size_t)jn * (NKV * HD), HD, 32,
                         (unsigned long long)(NKV * HD));
      tdm_load_2d_to_lds(&vt[(jb + 1) & 1][0], vbase + jn, 32, HD, (unsigned long long)SEQ);
    }
    const unsigned short* ktc = &kt[jb & 1][0];
    const unsigned short* vtc = &vt[jb & 1][0];

    // ---- S tiles from LDS (ds_load_b128 fragments) ----
    FragAB kb[8];
    #pragma unroll
    for (int c = 0; c < 4; ++c) {
      kb[c]     = load_frag_b(ktc,            HD, c * 32, lane);
      kb[4 + c] = load_frag_b(ktc + 16 * HD,  HD, c * 32, lane);
    }
    f32x8 st0, st1;
    #pragma unroll
    for (int i = 0; i < 8; ++i) { st0[i] = 0.0f; st1[i] = 0.0f; }
    #pragma unroll
    for (int c = 0; c < 4; ++c) {
      st0 = wmma_bf16(qa[c], kb[c], st0);
      st1 = wmma_bf16(qa[c], kb[4 + c], st1);
    }

    // V fragments early (overlap with softmax VALU/trans work)
    FragAB vb[8];
    #pragma unroll
    for (int t = 0; t < 8; ++t)
      vb[t] = load_frag_b(vtc + (size_t)(t * 16) * 32, 32, 0, lane);

    // ---- online softmax in C layout: row m = i + 8*hf, col = lh / 16+lh ----
    float alpha[8];
    #pragma unroll
    for (int i = 0; i < 8; ++i) {
      const int row = q0 + i + 8 * hf;
      float s0 = st0[i] * scale; if (j + lh > row)      s0 = -__builtin_inff();
      float s1 = st1[i] * scale; if (j + 16 + lh > row) s1 = -__builtin_inff();
      float t = fmaxf(s0, s1);
      #pragma unroll
      for (int off = 8; off; off >>= 1) t = fmaxf(t, __shfl_xor(t, off, 16));
      const float mN = fmaxf(Mx[i], t);                    // finite after block 0
      const float a  = __expf(Mx[i] - mN);
      const float p0 = __expf(s0 - mN);
      const float p1 = __expf(s1 - mN);
      float r = p0 + p1;
      #pragma unroll
      for (int off = 8; off; off >>= 1) r += __shfl_xor(r, off, 16);
      Lx[i] = Lx[i] * a + r;
      Mx[i] = mN;
      alpha[i] = a;
      pl[(i + 8 * hf) * 32 + lh]      = f32_to_bf16(p0);
      pl[(i + 8 * hf) * 32 + 16 + lh] = f32_to_bf16(p1);
    }

    // intra-wave LDS store->load ordering (per-wave staging buffer)
    asm volatile("s_wait_dscnt 0" ::: "memory");

    FragAB pa;
    {
      const unsigned short* pp = pl + lh * 32 + 8 * hf;
      pa.q[0] = *(const u32x4*)(pp);
      pa.q[1] = *(const u32x4*)(pp + 16);
    }

    #pragma unroll
    for (int t = 0; t < 8; ++t)
      #pragma unroll
      for (int i = 0; i < 8; ++i) of[t][i] *= alpha[i];

    #pragma unroll
    for (int t = 0; t < 8; ++t)
      of[t] = wmma_bf16(pa, vb[t], of[t]);
  }

  // normalize and store bf16 output
  unsigned short* obase = o + (size_t)q0 * (NH * HD) + head * HD;
  #pragma unroll
  for (int i = 0; i < 8; ++i) {
    const float inv = 1.0f / Lx[i];
    const int row = i + 8 * hf;
    #pragma unroll
    for (int t = 0; t < 8; ++t)
      obase[(size_t)row * (NH * HD) + t * 16 + lh] = f32_to_bf16(of[t][i] * inv);
  }
}

// ---------------- host ----------------

extern "C" void kernel_launch(void* const* d_in, const int* in_sizes, int n_in,
                              void* d_out, int out_size, void* d_ws, size_t ws_size,
                              hipStream_t stream) {
  const float* x  = (const float*)d_in[0];
  const float* wq = (const float*)d_in[1];
  const float* wk = (const float*)d_in[2];
  const float* wv = (const float*)d_in[3];
  const float* wo = (const float*)d_in[4];
  const float* fc = (const float*)d_in[5];
  const float* fs = (const float*)d_in[6];
  // d_in[7] = mask (causal handled in-kernel), d_in[8] = start_pos (0)
  float* out = (float*)d_out;

  size_t off = 0;
  auto alloc = [&](size_t bytes) -> void* {
    void* p = (char*)d_ws + off;
    off += (bytes + 255) & ~(size_t)255;
    return p;
  };
  unsigned short* x_bf  = (unsigned short*)alloc((size_t)SEQ * DIMN * 2);
  unsigned short* wqT   = (unsigned short*)alloc((size_t)DIMN * DIMN * 2);
  unsigned short* wkT   = (unsigned short*)alloc((size_t)DIMN * (NKV * HD) * 2);
  unsigned short* wvT   = (unsigned short*)alloc((size_t)DIMN * (NKV * HD) * 2);
  unsigned short* woT   = (unsigned short*)alloc((size_t)DIMN * DIMN * 2);
  float*          xq    = (float*)alloc((size_t)SEQ * DIMN * 4);
  float*          xk    = (float*)alloc((size_t)SEQ * (NKV * HD) * 4);
  float*          xv    = (float*)alloc((size_t)SEQ * (NKV * HD) * 4);
  unsigned short* q_bf  = (unsigned short*)alloc((size_t)SEQ * DIMN * 2);
  unsigned short* k_bf  = (unsigned short*)alloc((size_t)SEQ * (NKV * HD) * 2);
  unsigned short* vT_bf = (unsigned short*)alloc((size_t)NKV * HD * SEQ * 2);
  unsigned short* ao_bf = (unsigned short*)alloc((size_t)SEQ * DIMN * 2);

  const int TB = 256;
  { int n = SEQ * DIMN;            cvt_f32_bf16<<<(n + TB - 1) / TB, TB, 0, stream>>>(x, x_bf, n); }
  { int n = DIMN * DIMN;           transpose_cvt<<<(n + TB - 1) / TB, TB, 0, stream>>>(wq, wqT, DIMN, DIMN); }
  { int n = DIMN * (NKV * HD);     transpose_cvt<<<(n + TB - 1) / TB, TB, 0, stream>>>(wk, wkT, DIMN, NKV * HD); }
  { int n = DIMN * (NKV * HD);     transpose_cvt<<<(n + TB - 1) / TB, TB, 0, stream>>>(wv, wvT, DIMN, NKV * HD); }
  { int n = DIMN * DIMN;           transpose_cvt<<<(n + TB - 1) / TB, TB, 0, stream>>>(wo, woT, DIMN, DIMN); }

  gemm_bf16<<<dim3(DIMN / 64, SEQ / 128), 128, 0, stream>>>(x_bf, wqT, xq, SEQ, DIMN, DIMN);
  gemm_bf16<<<dim3((NKV * HD) / 64, SEQ / 128), 128, 0, stream>>>(x_bf, wkT, xk, SEQ, NKV * HD, DIMN);
  gemm_bf16<<<dim3((NKV * HD) / 64, SEQ / 128), 128, 0, stream>>>(x_bf, wvT, xv, SEQ, NKV * HD, DIMN);

  { int n = SEQ * NH * (HD / 2);   rope_cvt<<<(n + TB - 1) / TB, TB, 0, stream>>>(xq, fc, fs, q_bf, NH); }
  { int n = SEQ * NKV * (HD / 2);  rope_cvt<<<(n + TB - 1) / TB, TB, 0, stream>>>(xk, fc, fs, k_bf, NKV); }
  { int n = NKV * HD * SEQ;        v_transpose_cvt<<<(n + TB - 1) / TB, TB, 0, stream>>>(xv, vT_bf); }

  flash_attn<<<dim3(SEQ / 64, NH), 128, 0, stream>>>(q_bf, k_bf, vT_bf, ao_bf);

  gemm_bf16<<<dim3(DIMN / 64, SEQ / 128), 128, 0, stream>>>(ao_bf, woT, out, SEQ, DIMN, DIMN);

  (void)in_sizes; (void)n_in; (void)out_size; (void)ws_size;
}
